// HOLE_49649821942231
// MI455X (gfx1250) — compile-verified, matrix operand order checked by reading, not run on hardware
//
#include <hip/hip_runtime.h>
#include <hip/hip_bf16.h>
#include <math.h>

typedef __bf16 bf16_t;
typedef __attribute__((ext_vector_type(16))) __bf16 v16bf;
typedef __attribute__((ext_vector_type(8)))  __bf16 v8bf;
typedef __attribute__((ext_vector_type(8)))  float  v8f;

#define BATCH  1024
#define DIMS   512
#define KE     100000   // NUM_ENTITIES
#define KR     1000     // NUM_RELATIONS
#define KSPLIT 8
#define BLK_M  128      // per-workgroup M tile (8 waves x 16 rows)
#define BLK_N  128      // per-workgroup N tile (8 x 16-wide WMMA frags)
#define NFRAG  (BLK_N / 16)
#define LDS_STRIDE 72   // halves per LDS tile row (64 + 8 pad -> 36-dword
                        // lane stride: conflict-free b128 reads on 64 banks)

// ---------------------------------------------------------------------------
// WMMA helper: D = A(16x32 bf16) * B(32x16 bf16) + C(16x16 f32)
// ---------------------------------------------------------------------------
static __device__ __forceinline__ v8f wmma_bf16(v16bf a, v16bf b, v8f c) {
  return __builtin_amdgcn_wmma_f32_16x16x32_bf16(
      false, a, false, b, (short)0, c, false, false);
}

// ---------------------------------------------------------------------------
// Direct-global bf16 fragment load (A fragments + leftover/tail B fragments).
// CDNA5 16-bit frag layout (ISA 7.12.2, lane-symmetric for A/B): lane l<16
// holds row l, K in {kb..kb+7} U {kb+16..kb+23}, kb = k0 (+8 for lanes 16-31).
// ---------------------------------------------------------------------------
template <bool GUARD>
static __device__ __forceinline__ v16bf load_frag(const float* __restrict__ mat,
                                                  int row0, int rowStride,
                                                  int k0, int kMax) {
  const int lane = threadIdx.x & 31;
  const int r    = row0 + (lane & 15);
  const int kb   = k0 + ((lane & 16) >> 1);   // +0 or +8
  const float* p = mat + (size_t)r * rowStride + kb;
  v16bf f;
  if (!GUARD) {
    const float4* q = (const float4*)p;       // rows & k0 are 16B aligned
    float4 u0 = q[0];
    float4 u1 = q[1];
    float4 u2 = q[4];   // +16 floats
    float4 u3 = q[5];
    __builtin_prefetch(p + 64, 0, 1);         // next K pair (global_prefetch)
    f[0]  = (bf16_t)u0.x; f[1]  = (bf16_t)u0.y; f[2]  = (bf16_t)u0.z; f[3]  = (bf16_t)u0.w;
    f[4]  = (bf16_t)u1.x; f[5]  = (bf16_t)u1.y; f[6]  = (bf16_t)u1.z; f[7]  = (bf16_t)u1.w;
    f[8]  = (bf16_t)u2.x; f[9]  = (bf16_t)u2.y; f[10] = (bf16_t)u2.z; f[11] = (bf16_t)u2.w;
    f[12] = (bf16_t)u3.x; f[13] = (bf16_t)u3.y; f[14] = (bf16_t)u3.z; f[15] = (bf16_t)u3.w;
  } else {
    const float* row = mat + (size_t)r * rowStride;
#pragma unroll
    for (int i = 0; i < 8; ++i) {
      const int ka = kb + i;
      const int kc = kb + 16 + i;
      const int kaC = (ka < kMax) ? ka : (kMax - 1);  // clamp: stay in-bounds
      const int kcC = (kc < kMax) ? kc : (kMax - 1);
      const float va = row[kaC];
      const float vc = row[kcC];
      f[i]     = (bf16_t)((ka < kMax) ? va : 0.0f);
      f[8 + i] = (bf16_t)((kc < kMax) ? vc : 0.0f);
    }
  }
  return f;
}

// ---------------------------------------------------------------------------
// Cooperative staging of a 128(N) x 64(K) f32 tile -> bf16 LDS tile.
// Thread t handles row t/2, half-row (t&1)*32: 8x global_load_b128,
// 16x v_cvt_pk_bf16_f32, 4x ds_store_b128.
// ---------------------------------------------------------------------------
static __device__ __forceinline__ void tile_load(const float* __restrict__ W,
                                                 int n0, int rowStride, int k0,
                                                 float4 (&u)[8]) {
  const int t   = threadIdx.x;
  const int row = t >> 1;
  const int kh  = (t & 1) * 32;
  const float4* q =
      (const float4*)(W + (size_t)(n0 + row) * rowStride + k0 + kh);
#pragma unroll
  for (int i = 0; i < 8; ++i) u[i] = q[i];
}

static __device__ __forceinline__ void tile_cvt_store(const float4 (&u)[8],
                                                      bf16_t* __restrict__ lds) {
  const int t   = threadIdx.x;
  const int row = t >> 1;
  const int kh  = (t & 1) * 32;
  bf16_t* dst = lds + row * LDS_STRIDE + kh;
#pragma unroll
  for (int h = 0; h < 4; ++h) {
    v8bf a;
    a[0] = (bf16_t)u[2 * h].x;     a[1] = (bf16_t)u[2 * h].y;
    a[2] = (bf16_t)u[2 * h].z;     a[3] = (bf16_t)u[2 * h].w;
    a[4] = (bf16_t)u[2 * h + 1].x; a[5] = (bf16_t)u[2 * h + 1].y;
    a[6] = (bf16_t)u[2 * h + 1].z; a[7] = (bf16_t)u[2 * h + 1].w;
    *(v8bf*)(dst + 8 * h) = a;
  }
}

// B fragment from staged LDS tile: two ds_load_b128 per lane.
static __device__ __forceinline__ v16bf lds_frag(const bf16_t* __restrict__ tile,
                                                 int nloc, int ks) {
  const int lane = threadIdx.x & 31;
  const int row  = nloc + (lane & 15);
  const int kb   = ks + ((lane & 16) >> 1);
  const bf16_t* p = tile + row * LDS_STRIDE + kb;
  v8bf lo = *(const v8bf*)p;          // K kb..kb+7
  v8bf hi = *(const v8bf*)(p + 16);   // K kb+16..kb+23
  return __builtin_shufflevector(lo, hi, 0, 1, 2, 3, 4, 5, 6, 7,
                                 8, 9, 10, 11, 12, 13, 14, 15);
}

// Store a 16x16 f32 C/D fragment (VGPR g -> M = g + 8*(lane>=16), N = lane&15)
static __device__ __forceinline__ void store_frag(float* __restrict__ dst,
                                                  int ldd, const v8f& c) {
  const int lane = threadIdx.x & 31;
  const int cn   = lane & 15;
  const int rb   = (lane >> 4) << 3;
#pragma unroll
  for (int g = 0; g < 8; ++g) dst[(size_t)(rb + g) * ldd + cn] = c[g];
}

// ---------------------------------------------------------------------------
// Kernel 1: fused split-K GEMM  A = x @ We^T, B = y @ We^T  (bf16 WMMA).
// 128-wide N tile staged once per WG as bf16 in double-buffered LDS; each
// B fragment feeds 2 WMMAs (x and y); each A fragment feeds 8 WMMAs.
// Grid: (8, 4, KSPLIT); block = 256 (8 waves).
// ---------------------------------------------------------------------------
__global__ __launch_bounds__(256) void gemm_xy_kernel(
    const float* __restrict__ x, const float* __restrict__ y,
    const float* __restrict__ We, float* __restrict__ Apart,
    float* __restrict__ Bpart) {
  __shared__ __align__(16) bf16_t tileW[2][BLK_N * LDS_STRIDE];

  const int wave  = threadIdx.x >> 5;
  const int m0    = blockIdx.x * BLK_M + wave * 16;
  const int n0    = blockIdx.y * BLK_N;
  const int chunk = blockIdx.z;

  const int totalSteps = KE / 32;                       // 3125 (odd!)
  const int sBeg = (totalSteps * chunk) / KSPLIT;
  const int sEnd = (totalSteps * (chunk + 1)) / KSPLIT;

  v8f accX[NFRAG] = {};
  v8f accY[NFRAG] = {};

  int s   = sBeg;
  int buf = 0;
  if (s + 1 < sEnd) {  // prologue: stage first K-pair tile
    float4 u[8];
    tile_load(We, n0, KE, s * 32, u);
    tile_cvt_store(u, &tileW[0][0]);
  }
  __syncthreads();

  for (; s + 1 < sEnd; s += 2) {
    const bool hasNext = (s + 3) < sEnd;
    float4 u[8];
    if (hasNext)  // issue next tile's global loads early
      tile_load(We, n0, KE, (s + 2) * 32, u);

    const v16bf ax0 = load_frag<false>(x, m0, KE, s * 32, KE);
    const v16bf ax1 = load_frag<false>(x, m0, KE, s * 32 + 32, KE);
    const v16bf ay0 = load_frag<false>(y, m0, KE, s * 32, KE);
    const v16bf ay1 = load_frag<false>(y, m0, KE, s * 32 + 32, KE);

#pragma unroll
    for (int j = 0; j < NFRAG; ++j) {
      const v16bf b0 = lds_frag(&tileW[buf][0], j * 16, 0);
      const v16bf b1 = lds_frag(&tileW[buf][0], j * 16, 32);
      accX[j] = wmma_bf16(ax0, b0, accX[j]);
      accY[j] = wmma_bf16(ay0, b0, accY[j]);
      accX[j] = wmma_bf16(ax1, b1, accX[j]);
      accY[j] = wmma_bf16(ay1, b1, accY[j]);
    }

    if (hasNext) tile_cvt_store(u, &tileW[buf ^ 1][0]);
    __syncthreads();
    buf ^= 1;
  }

  if (s < sEnd) {  // odd leftover 32-step: direct-global B fragments
    const int k0 = s * 32;
    const v16bf ax = load_frag<false>(x, m0, KE, k0, KE);
    const v16bf ay = load_frag<false>(y, m0, KE, k0, KE);
#pragma unroll
    for (int j = 0; j < NFRAG; ++j) {
      const v16bf bw = load_frag<false>(We, n0 + j * 16, KE, k0, KE);
      accX[j] = wmma_bf16(ax, bw, accX[j]);
      accY[j] = wmma_bf16(ay, bw, accY[j]);
    }
  }

  float* pa = Apart + ((size_t)chunk * BATCH + m0) * DIMS + n0;
  float* pb = Bpart + ((size_t)chunk * BATCH + m0) * DIMS + n0;
#pragma unroll
  for (int j = 0; j < NFRAG; ++j) {
    store_frag(pa + j * 16, DIMS, accX[j]);
    store_frag(pb + j * 16, DIMS, accY[j]);
  }
}

// ---------------------------------------------------------------------------
// Kernel 2: RR = r @ Wr^T.  K = 1000 = 15 staged pairs (960) + one full
// direct 32-step + one guarded 8-wide tail. Grid (8, 4).
// ---------------------------------------------------------------------------
__global__ __launch_bounds__(256) void gemm_r_kernel(
    const float* __restrict__ r, const float* __restrict__ Wr,
    float* __restrict__ RR) {
  __shared__ __align__(16) bf16_t tileW[2][BLK_N * LDS_STRIDE];

  const int wave = threadIdx.x >> 5;
  const int m0   = blockIdx.x * BLK_M + wave * 16;
  const int n0   = blockIdx.y * BLK_N;

  v8f acc[NFRAG] = {};
  const int sEnd = KR / 32;  // 31 full 32-steps

  int s   = 0;
  int buf = 0;
  {
    float4 u[8];
    tile_load(Wr, n0, KR, 0, u);
    tile_cvt_store(u, &tileW[0][0]);
  }
  __syncthreads();

  for (; s + 1 < sEnd; s += 2) {
    const bool hasNext = (s + 3) < sEnd;
    float4 u[8];
    if (hasNext) tile_load(Wr, n0, KR, (s + 2) * 32, u);

    const v16bf ar0 = load_frag<false>(r, m0, KR, s * 32, KR);
    const v16bf ar1 = load_frag<false>(r, m0, KR, s * 32 + 32, KR);
#pragma unroll
    for (int j = 0; j < NFRAG; ++j) {
      const v16bf b0 = lds_frag(&tileW[buf][0], j * 16, 0);
      const v16bf b1 = lds_frag(&tileW[buf][0], j * 16, 32);
      acc[j] = wmma_bf16(ar0, b0, acc[j]);
      acc[j] = wmma_bf16(ar1, b1, acc[j]);
    }

    if (hasNext) tile_cvt_store(u, &tileW[buf ^ 1][0]);
    __syncthreads();
    buf ^= 1;
  }

  if (s < sEnd) {  // s = 30: full 32-step, direct global (k 960..991 < 1000)
    const int k0 = s * 32;
    const v16bf ar = load_frag<false>(r, m0, KR, k0, KR);
#pragma unroll
    for (int j = 0; j < NFRAG; ++j) {
      const v16bf bw = load_frag<false>(Wr, n0 + j * 16, KR, k0, KR);
      acc[j] = wmma_bf16(ar, bw, acc[j]);
    }
  }
  {  // guarded tail: k0 = 992, 8 valid K columns, zero-padded clamped loads
    const int k0 = sEnd * 32;
    const v16bf ar = load_frag<true>(r, m0, KR, k0, KR);
#pragma unroll
    for (int j = 0; j < NFRAG; ++j) {
      const v16bf bw = load_frag<true>(Wr, n0 + j * 16, KR, k0, KR);
      acc[j] = wmma_bf16(ar, bw, acc[j]);
    }
  }

  float* prr = RR + (size_t)m0 * DIMS + n0;
#pragma unroll
  for (int j = 0; j < NFRAG; ++j) store_frag(prr + j * 16, DIMS, acc[j]);
}

// ---------------------------------------------------------------------------
// Kernel 3: per batch row — sum split-K partials, then
//   score = (1/d) * sum_j a[j]*conv[j],  conv[j] = sum_m b[m]*rr[(j-m)&511]
// then sigmoid. One workgroup per row; conflict-free LDS pattern.
// ---------------------------------------------------------------------------
__global__ __launch_bounds__(256) void score_kernel(
    const float* __restrict__ Apart, const float* __restrict__ Bpart,
    const float* __restrict__ RR, float* __restrict__ out) {
  __shared__ float sa[DIMS], sb[DIMS], sr[DIMS];
  __shared__ float red[256];
  const int i   = blockIdx.x;
  const int tid = threadIdx.x;

  for (int j = tid; j < DIMS; j += 256) {
    float va = 0.0f, vb = 0.0f;
#pragma unroll
    for (int c = 0; c < KSPLIT; ++c) {
      const size_t o = ((size_t)c * BATCH + i) * DIMS + j;
      va += Apart[o];
      vb += Bpart[o];
    }
    sa[j] = va;
    sb[j] = vb;
    sr[j] = RR[(size_t)i * DIMS + j];
  }
  __syncthreads();

  float local = 0.0f;
#pragma unroll
  for (int t = 0; t < DIMS / 256; ++t) {
    const int j = tid + t * 256;
    float conv = 0.0f;
    for (int m = 0; m < DIMS; ++m)
      conv = fmaf(sb[m], sr[(j - m) & (DIMS - 1)], conv);
    local = fmaf(sa[j], conv, local);
  }

  red[tid] = local;
  __syncthreads();
  for (int sft = 128; sft > 0; sft >>= 1) {
    if (tid < sft) red[tid] += red[tid + sft];
    __syncthreads();
  }
  if (tid == 0) {
    const float sc = red[0] * (1.0f / (float)DIMS);
    out[i] = 1.0f / (1.0f + expf(-sc));
  }
}

// ---------------------------------------------------------------------------
// Launcher. Workspace (f32): Apart[8][1024][512] | Bpart[8][1024][512] |
// RR[1024][512] = 34 MB. Every slot fully rewritten each call (deterministic).
// ---------------------------------------------------------------------------
extern "C" void kernel_launch(void* const* d_in, const int* in_sizes, int n_in,
                              void* d_out, int out_size, void* d_ws,
                              size_t ws_size, hipStream_t stream) {
  const float* x  = (const float*)d_in[0];
  const float* y  = (const float*)d_in[1];
  const float* r  = (const float*)d_in[2];
  const float* We = (const float*)d_in[3];
  const float* Wr = (const float*)d_in[4];
  float* out = (float*)d_out;

  float* Apart = (float*)d_ws;
  float* Bpart = Apart + (size_t)KSPLIT * BATCH * DIMS;
  float* RR    = Bpart + (size_t)KSPLIT * BATCH * DIMS;

  dim3 g1(BATCH / BLK_M, DIMS / BLK_N, KSPLIT);   // 8 x 4 x 8 = 256 WGs
  gemm_xy_kernel<<<g1, 256, 0, stream>>>(x, y, We, Apart, Bpart);

  dim3 g2(BATCH / BLK_M, DIMS / BLK_N, 1);        // 32 WGs
  gemm_r_kernel<<<g2, 256, 0, stream>>>(r, Wr, RR);

  score_kernel<<<BATCH, 256, 0, stream>>>(Apart, Bpart, RR, out);
}
